// TargetMLPReadout_5368709120481
// MI455X (gfx1250) — matrix operand order repaired and checked by reading, not compiled, not run on hardware
//
#include <hip/hip_runtime.h>

typedef __attribute__((ext_vector_type(16))) _Float16 v16h;
typedef __attribute__((ext_vector_type(8)))  _Float16 v8h;
typedef __attribute__((ext_vector_type(8)))  float    v8f;
typedef __attribute__((ext_vector_type(4)))  float    v4f;

#define BATCH   256
#define NNODE   2048
#define DIM     128
#define NTROW   2047                 // non-target rows per batch
#define RTOT    (BATCH * NTROW)      // 524032 rows total
#define MT      64                   // rows per tile
#define NTILES  (RTOT / MT)          // 8188 (exact)
#define NBLK    256
#define THREADS 256

__global__ void zero_out_kernel(float* __restrict__ out) {
  int i = blockIdx.x * blockDim.x + threadIdx.x;
  if (i < BATCH * DIM) out[i] = 0.0f;
}

__global__ void reduce_partials_kernel(const float* __restrict__ part,
                                       float* __restrict__ out, int nblk) {
  int i = blockIdx.x * blockDim.x + threadIdx.x;
  if (i >= BATCH * DIM) return;
  float s = 0.0f;
  for (int b = 0; b < nblk; ++b) s += part[(size_t)b * (BATCH * DIM) + i];
  out[i] = s;
}

// Stage one 64-row x tile (rows = [target emb | node emb], fp32 -> f16) into
// xbuf, and the 64 segment ids into segbuf (async DMA to LDS, ASYNCcnt path).
__device__ __forceinline__ void stage_tile(
    const float* __restrict__ embs, const int* __restrict__ batch_idx,
    _Float16* __restrict__ xbuf, int* __restrict__ segbuf,
    int tile, int tid, int stride)
{
  const int xrow = tid >> 2;   // 0..63
  const int q    = tid & 3;    // quarter of the 256-wide x row
  int gr = tile * MT + xrow;
  int bb = gr / NTROW;
  int nn = 1 + gr % NTROW;
  int srow = (q < 2) ? (bb * NNODE) : (bb * NNODE + nn);
  const v4f* src = (const v4f*)(embs + srow * DIM + (q & 1) * 64);
  v16h* dst = (v16h*)&xbuf[xrow * 256 + q * 64];
  #pragma unroll
  for (int i = 0; i < 4; ++i) {
    v4f f0 = src[i * 4 + 0], f1 = src[i * 4 + 1];
    v4f f2 = src[i * 4 + 2], f3 = src[i * 4 + 3];
    v16h h;
    #pragma unroll
    for (int e = 0; e < 4; ++e) {
      h[e]      = (_Float16)f0[e];
      h[4 + e]  = (_Float16)f1[e];
      h[8 + e]  = (_Float16)f2[e];
      h[12 + e] = (_Float16)f3[e];
    }
    dst[i] = h;
  }
  if (tid < MT) {
    int gr2 = tile * MT + tid;
    int bi  = gr2 / NTROW;
    int nn2 = 1 + gr2 % NTROW;
    const int* gsrc = batch_idx + bi * NNODE + nn2;
    unsigned ldsoff = (unsigned)(size_t)(void*)&segbuf[tid];
    // async DMA: 4B per lane, global -> LDS, tracked by ASYNCcnt
    asm volatile("global_load_async_to_lds_b32 %0, %1, off"
                 :: "v"(ldsoff), "v"(gsrc) : "memory");
  }
  // pull the tile after next into L2 (global_prefetch_b8)
  int pt = tile + stride;
  if (pt < NTILES) {
    int pg = pt * MT + xrow;
    int pb = pg / NTROW;
    int pn = 1 + pg % NTROW;
    int ps = (q < 2) ? (pb * NNODE) : (pb * NNODE + pn);
    __builtin_prefetch(embs + ps * DIM + (q & 1) * 64, 0, 0);
  }
}

__global__ __launch_bounds__(THREADS) void mlp_seg_kernel(
    const float* __restrict__ embs, const float* __restrict__ W1,
    const float* __restrict__ b1,   const float* __restrict__ W2,
    const float* __restrict__ b2,   const int* __restrict__ batch_idx,
    float* __restrict__ out,        float* __restrict__ partials)
{
  __shared__ float    segAcc[BATCH * DIM];    // 128 KB persistent segment accumulator
  __shared__ _Float16 w1p[8 * 8 * 32 * 16];   //  64 KB W1 in B-fragment order
  __shared__ _Float16 w2p[4 * 8 * 32 * 16];   //  32 KB W2 in B-fragment order
  __shared__ _Float16 xT[2][MT * 256];        //  64 KB double-buffered x tiles (f16)
  __shared__ _Float16 hTile[MT * 128];        //  16 KB hidden rows (f16)
  __shared__ int      segT[2][MT];

  const int tid  = threadIdx.x;
  const int lane = tid & 31;
  const int wave = tid >> 5;
  const int ln   = lane & 15;       // N position / A-matrix M row
  const int g    = lane >> 4;       // lane group (K offset selector)
  const int rg   = wave >> 1;       // row-group of 16 rows (0..3)
  const int ch   = wave & 1;        // column half (0..1)
  const int rowBase = rg * 16;
  const int colBase = ch * 64;
  const int ntBase  = ch * 4;

  // ---- pack W1 (128x256) as f16 B-fragments: slot=(kc*8+nt)*32+lane holds
  //      B[kc*32 + 16*g + i][nt*16 + ln] = W1[n][k], 16 contiguous k per slot ----
  for (int s = tid; s < 8 * 8 * 32; s += THREADS) {
    int kc = s >> 8, nt = (s >> 5) & 7, l = s & 31;
    int n  = nt * 16 + (l & 15);
    int kb = kc * 32 + (l >> 4) * 16;
    const float* w = W1 + n * 256 + kb;
    v16h v;
    #pragma unroll
    for (int i = 0; i < 16; ++i) v[i] = (_Float16)w[i];
    *(v16h*)&w1p[s * 16] = v;
  }
  // ---- pack W2 (128x128) likewise ----
  for (int s = tid; s < 4 * 8 * 32; s += THREADS) {
    int kc = s >> 8, nt = (s >> 5) & 7, l = s & 31;
    int n  = nt * 16 + (l & 15);
    int kb = kc * 32 + (l >> 4) * 16;
    const float* w = W2 + n * 128 + kb;
    v16h v;
    #pragma unroll
    for (int i = 0; i < 16; ++i) v[i] = (_Float16)w[i];
    *(v16h*)&w2p[s * 16] = v;
  }
  for (int i = tid; i < BATCH * DIM; i += THREADS) segAcc[i] = 0.0f;

  float b1v[4], b2v[4];
  #pragma unroll
  for (int t = 0; t < 4; ++t) {
    b1v[t] = b1[colBase + t * 16 + ln];
    b2v[t] = b2[colBase + t * 16 + ln];
  }

  const int stride = (int)gridDim.x;
  int p = 0;
  if (blockIdx.x < NTILES)
    stage_tile(embs, batch_idx, xT[0], segT[0], blockIdx.x, tid, stride);

  for (int tile = blockIdx.x; tile < NTILES; tile += stride) {
    // publish buf[p]: async seg DMA must land before the barrier
    asm volatile("s_wait_asynccnt 0x0" ::: "memory");
    __syncthreads();

    // overlap: stage the next tile into the other buffer under the GEMMs
    int nxt = tile + stride;
    if (nxt < NTILES)
      stage_tile(embs, batch_idx, xT[p ^ 1], segT[p ^ 1], nxt, tid, stride);

    // ---- GEMM1: h[16x64] = x[16x256] * W1^T, f16 WMMA, f32 accumulate ----
    const _Float16* xcur = xT[p];
    v8f c1[4] = {};
    #pragma unroll
    for (int kc = 0; kc < 8; ++kc) {
      const _Float16* ap = &xcur[(rowBase + ln) * 256 + kc * 32 + g * 16];
      v8h lo = *(const v8h*)ap;
      v8h hi = *(const v8h*)(ap + 8);
      v16h a = __builtin_shufflevector(lo, hi, 0, 1, 2, 3, 4, 5, 6, 7,
                                                8, 9, 10, 11, 12, 13, 14, 15);
      #pragma unroll
      for (int t = 0; t < 4; ++t) {
        v16h bm = *(const v16h*)&w1p[((kc * 8 + ntBase + t) * 32 + lane) * 16];
        c1[t] = __builtin_amdgcn_wmma_f32_16x16x32_f16(
            false, a, false, bm, (short)0, c1[t], false, false);
      }
    }
    // bias + ReLU, spill hidden rows to LDS as f16 (C layout: M = j + 8*g)
    #pragma unroll
    for (int t = 0; t < 4; ++t) {
      int n = colBase + t * 16 + ln;
      #pragma unroll
      for (int j = 0; j < 8; ++j) {
        float v = c1[t][j] + b1v[t];
        hTile[(rowBase + j + 8 * g) * 128 + n] = (_Float16)(v > 0.0f ? v : 0.0f);
      }
    }
    __syncthreads();

    // ---- GEMM2: y[16x64] = h[16x128] * W2^T ----
    v8f c2[4] = {};
    #pragma unroll
    for (int kc = 0; kc < 4; ++kc) {
      const _Float16* ap = &hTile[(rowBase + ln) * 128 + kc * 32 + g * 16];
      v8h lo = *(const v8h*)ap;
      v8h hi = *(const v8h*)(ap + 8);
      v16h a = __builtin_shufflevector(lo, hi, 0, 1, 2, 3, 4, 5, 6, 7,
                                                8, 9, 10, 11, 12, 13, 14, 15);
      #pragma unroll
      for (int t = 0; t < 4; ++t) {
        v16h bm = *(const v16h*)&w2p[((kc * 8 + ntBase + t) * 32 + lane) * 16];
        c2[t] = __builtin_amdgcn_wmma_f32_16x16x32_f16(
            false, a, false, bm, (short)0, c2[t], false, false);
      }
    }
    // ---- segment-accumulate y (+b2) into the LDS accumulator (ds_add_f32) ----
    const int* segcur = segT[p];
    #pragma unroll
    for (int t = 0; t < 4; ++t) {
      int n = colBase + t * 16 + ln;
      #pragma unroll
      for (int j = 0; j < 8; ++j) {
        int r = rowBase + j + 8 * g;
        atomicAdd(&segAcc[segcur[r] * DIM + n], c2[t][j] + b2v[t]);
      }
    }
    p ^= 1;
  }

  __syncthreads();
  // ---- flush: plain stores into workspace if available, else global fp32 atomics ----
  if (partials) {
    float* dst = partials + (size_t)blockIdx.x * (BATCH * DIM);
    for (int i = tid; i < BATCH * DIM; i += THREADS) dst[i] = segAcc[i];
  } else {
    for (int i = tid; i < BATCH * DIM; i += THREADS)
      unsafeAtomicAdd(&out[i], segAcc[i]);
  }
}

extern "C" void kernel_launch(void* const* d_in, const int* in_sizes, int n_in,
                              void* d_out, int out_size, void* d_ws, size_t ws_size,
                              hipStream_t stream) {
  const float* embs      = (const float*)d_in[0];
  const float* W1        = (const float*)d_in[1];
  const float* b1        = (const float*)d_in[2];
  const float* W2        = (const float*)d_in[3];
  const float* b2        = (const float*)d_in[4];
  const int*   batch_idx = (const int*)d_in[5];
  float* out = (float*)d_out;

  const size_t part_bytes = (size_t)NBLK * (BATCH * DIM) * sizeof(float);
  const bool use_ws = (ws_size >= part_bytes);
  float* partials = use_ws ? (float*)d_ws : nullptr;

  if (!use_ws) {
    zero_out_kernel<<<(BATCH * DIM + 255) / 256, 256, 0, stream>>>(out);
  }
  mlp_seg_kernel<<<NBLK, THREADS, 0, stream>>>(embs, W1, b1, W2, b2,
                                               batch_idx, out, partials);
  if (use_ws) {
    reduce_partials_kernel<<<(BATCH * DIM + 255) / 256, 256, 0, stream>>>(
        partials, out, NBLK);
  }
}
